// GNN_11605001634347
// MI455X (gfx1250) — matrix operand order repaired and checked by reading, not compiled, not run on hardware
//
#include <hip/hip_runtime.h>

// ---------------------------------------------------------------------------
// 2-layer relational GAT (GNN) for MI455X / gfx1250, wave32.
// Dense 128x128 transforms: V_WMMA_F32_16X16X32_BF16 (f32 accumulate).
// Edge-sized tensors (hh, rh, inter-layer h) stored in bf16 to halve the
// dominant HBM traffic (workload is memory-bound: ~14 FLOP/byte).
// ---------------------------------------------------------------------------

typedef __attribute__((ext_vector_type(16))) __bf16 v16bf;
typedef __attribute__((ext_vector_type(8)))  float  v8f;

#define DIMD 128
#define BS_STRIDE 136   // 128 + 8 pad: row stride 272B (16B multiple), low bank conflict

// float -> monotonically ordered unsigned key (for atomicMax-based segment max)
__device__ __forceinline__ unsigned f2key(float f) {
    unsigned b = __float_as_uint(f);
    return (b & 0x80000000u) ? ~b : (b | 0x80000000u);
}
__device__ __forceinline__ float key2f(unsigned k) {
    unsigned b = (k & 0x80000000u) ? (k ^ 0x80000000u) : ~k;
    return __uint_as_float(b);
}
#define NEG_INF_KEY 0x007FFFFFu   // f2key(-inf)

// ---------------------------------------------------------------------------
// A-fragment loader: ISA 16-bit A layout (16x32), lane half h:
//   VGPR0..3 hold k = k0 + {0..7},  VGPR4..7 hold k = k0 + 16 + {0..7}
// where k0 = kt*32 + h*8.  For bf16 rows both k-runs are contiguous 16B.
// ---------------------------------------------------------------------------
__device__ __forceinline__ v16bf load_a_frag(const float* Arow, int k0) {
    float4 f0 = *(const float4*)(Arow + k0);
    float4 f1 = *(const float4*)(Arow + k0 + 4);
    float4 f2 = *(const float4*)(Arow + k0 + 16);
    float4 f3 = *(const float4*)(Arow + k0 + 20);
    v16bf a;
    a[0]  = (__bf16)f0.x; a[1]  = (__bf16)f0.y; a[2]  = (__bf16)f0.z; a[3]  = (__bf16)f0.w;
    a[4]  = (__bf16)f1.x; a[5]  = (__bf16)f1.y; a[6]  = (__bf16)f1.z; a[7]  = (__bf16)f1.w;
    a[8]  = (__bf16)f2.x; a[9]  = (__bf16)f2.y; a[10] = (__bf16)f2.z; a[11] = (__bf16)f2.w;
    a[12] = (__bf16)f3.x; a[13] = (__bf16)f3.y; a[14] = (__bf16)f3.z; a[15] = (__bf16)f3.w;
    return a;
}
__device__ __forceinline__ v16bf load_a_frag(const __bf16* Arow, int k0) {
    v16bf a;
    ((uint4*)&a)[0] = *(const uint4*)(Arow + k0);        // k0..k0+7
    ((uint4*)&a)[1] = *(const uint4*)(Arow + k0 + 16);   // k0+16..k0+23
    return a;
}

// ---------------------------------------------------------------------------
// C = A[rows,128] @ B[128,128]  (bf16 WMMA, f32 accum)
// FUSE: C = relu(A@B + addend)
// Block: 256 threads = 8 waves; each wave computes a 16x128 output strip.
// ---------------------------------------------------------------------------
template <typename TIN, typename TOUT, bool FUSE>
__global__ __launch_bounds__(256)
void gemm_wmma(const TIN* __restrict__ A,
               const float* __restrict__ Bw,       // [K=128][N=128] row-major f32
               const float* __restrict__ addend,   // [rows,128] f32 (FUSE only)
               TOUT* __restrict__ Cout,
               int rows)
{
    __shared__ __bf16 Bs[DIMD * BS_STRIDE];   // transposed: Bs[n][k]

    const int tid = threadIdx.x;

    // Cooperative load+transpose+convert of the weight tile: Bs[n*S + k] = B[k][n]
    for (int idx = tid; idx < DIMD * DIMD; idx += 256) {
        int k = idx >> 7;
        int n = idx & 127;
        Bs[n * BS_STRIDE + k] = (__bf16)Bw[idx];
    }
    __syncthreads();

    const int wave = tid >> 5;            // 0..7
    const int lane = tid & 31;
    const int m    = lane & 15;           // row-within-tile / col-within-tile
    const int half = lane >> 4;           // 0 or 1
    const int row0 = blockIdx.x * 128 + wave * 16;

    // Clamp OOB A rows to row 0 (loads stay valid; results masked at store)
    int arow = row0 + m;
    if (arow >= rows) arow = 0;
    const TIN* Arow = A + (size_t)arow * DIMD;

    v8f acc[8];
#pragma unroll
    for (int nt = 0; nt < 8; ++nt)
#pragma unroll
        for (int r = 0; r < 8; ++r) acc[nt][r] = 0.0f;

#pragma unroll
    for (int kt = 0; kt < 4; ++kt) {
        const v16bf a = load_a_frag(Arow, kt * 32 + half * 8);

        // B fragments: lane n = nt*16 + m, k = kt*32 + half*8 + {0..7, 16..23}
        // contiguous in the transposed LDS tile -> two ds_load_b128 each
        const int ks = kt * 32 + half * 8;
#pragma unroll
        for (int nt = 0; nt < 8; ++nt) {
            const __bf16* bp = &Bs[(nt * 16 + m) * BS_STRIDE + ks];
            v16bf b;
            ((uint4*)&b)[0] = *(const uint4*)(bp);
            ((uint4*)&b)[1] = *(const uint4*)(bp + 16);
            acc[nt] = __builtin_amdgcn_wmma_f32_16x16x32_bf16(
                false, a, false, b, (short)0, acc[nt], false, false);
        }
    }

    // Store: C layout -> VGPR r: lanes 0-15 M=r, lanes 16-31 M=r+8; N=lane&15
    const int crow0 = row0 + half * 8;
#pragma unroll
    for (int nt = 0; nt < 8; ++nt) {
        const int col = nt * 16 + m;
#pragma unroll
        for (int r = 0; r < 8; ++r) {
            const int rr = crow0 + r;
            if (rr < rows) {
                float v = acc[nt][r];
                if (FUSE) {
                    v += addend[(size_t)rr * DIMD + col];
                    v = v > 0.0f ? v : 0.0f;
                }
                Cout[(size_t)rr * DIMD + col] = (TOUT)v;
            }
        }
    }
}

// ---------------------------------------------------------------------------
// f32 -> bf16 bulk convert (n multiple of 4)
// ---------------------------------------------------------------------------
__global__ __launch_bounds__(256)
void f32_to_bf16(const float* __restrict__ in, __bf16* __restrict__ out, long n)
{
    long i = ((long)blockIdx.x * 256 + threadIdx.x) * 4;
    if (i >= n) return;
    float4 f = *(const float4*)(in + i);
    out[i]     = (__bf16)f.x;
    out[i + 1] = (__bf16)f.y;
    out[i + 2] = (__bf16)f.z;
    out[i + 3] = (__bf16)f.w;
}

// ---------------------------------------------------------------------------
// Init per-layer segment state: agg = 0, m_enc = key(-inf), s = 0
// ---------------------------------------------------------------------------
__global__ __launch_bounds__(256)
void init_seg(unsigned* __restrict__ menc, float* __restrict__ ssum,
              float* __restrict__ agg, int n)
{
    int i = blockIdx.x * blockDim.x + threadIdx.x;
    if (i < n * DIMD) agg[i] = 0.0f;
    if (i < n) { menc[i] = NEG_INF_KEY; ssum[i] = 0.0f; }
}

// ---------------------------------------------------------------------------
// nodeterm[n] = dot(hh[n,:], att2[:])   (warp per node)
// ---------------------------------------------------------------------------
__global__ __launch_bounds__(256)
void node_dot(const __bf16* __restrict__ hh, const float* __restrict__ att2,
              float* __restrict__ nt, int n)
{
    int node = (blockIdx.x * blockDim.x + threadIdx.x) >> 5;
    int lane = threadIdx.x & 31;
    if (node >= n) return;
    const __bf16* hr = hh + (size_t)node * DIMD;
    float p = 0.0f;
#pragma unroll
    for (int i = 0; i < 4; ++i) {
        int idx = lane + i * 32;
        p += (float)hr[idx] * att2[idx];
    }
    for (int off = 16; off; off >>= 1) p += __shfl_xor(p, off, 32);
    if (lane == 0) nt[node] = p;
}

// ---------------------------------------------------------------------------
// e = leaky(dot(hh[src],a1) + nodeterm[dst] + dot(rh,a3)); segment max via
// ordered-key atomicMax.  (warp per edge)
// ---------------------------------------------------------------------------
__global__ __launch_bounds__(256)
void edge_score(const __bf16* __restrict__ hh, const __bf16* __restrict__ rh,
                const int* __restrict__ src, const int* __restrict__ dst,
                const float* __restrict__ att,        // [384]: a1 | a2 | a3
                const float* __restrict__ ndot,
                float* __restrict__ eout, unsigned* __restrict__ menc, int ne)
{
    int edge = (blockIdx.x * blockDim.x + threadIdx.x) >> 5;
    int lane = threadIdx.x & 31;
    if (edge >= ne) return;
    const __bf16* hs = hh + (size_t)src[edge] * DIMD;
    const __bf16* rr = rh + (size_t)edge * DIMD;
    float p = 0.0f;
#pragma unroll
    for (int i = 0; i < 4; ++i) {
        int idx = lane + i * 32;
        p += (float)hs[idx] * att[idx] + (float)rr[idx] * att[256 + idx];
    }
    for (int off = 16; off; off >>= 1) p += __shfl_xor(p, off, 32);
    if (lane == 0) {
        int dd = dst[edge];
        float e = p + ndot[dd];
        e = e > 0.0f ? e : 0.01f * e;        // leaky_relu
        eout[edge] = e;
        atomicMax(&menc[dd], f2key(e));
    }
}

// ---------------------------------------------------------------------------
// ex = exp(e - m[dst]); s[dst] += ex  (thread per edge)
// ---------------------------------------------------------------------------
__global__ __launch_bounds__(256)
void edge_exp(float* __restrict__ e, const int* __restrict__ dst,
              const unsigned* __restrict__ menc, float* __restrict__ ssum, int ne)
{
    int i = blockIdx.x * blockDim.x + threadIdx.x;
    if (i >= ne) return;
    int dd = dst[i];
    float ex = __expf(e[i] - key2f(menc[dd]));
    e[i] = ex;
    atomicAdd(&ssum[dd], ex);
}

// ---------------------------------------------------------------------------
// agg[dst] += (ex/s[dst]) * (hh[src] + rh)   (warp per edge, f32 atomics)
// ---------------------------------------------------------------------------
__global__ __launch_bounds__(256)
void edge_agg(const __bf16* __restrict__ hh, const __bf16* __restrict__ rh,
              const int* __restrict__ src, const int* __restrict__ dst,
              const float* __restrict__ ex, const float* __restrict__ ssum,
              float* __restrict__ agg, int ne)
{
    int edge = (blockIdx.x * blockDim.x + threadIdx.x) >> 5;
    int lane = threadIdx.x & 31;
    if (edge >= ne) return;
    int ss = src[edge];
    int dd = dst[edge];
    float alpha = ex[edge] / ssum[dd];
    const __bf16* hs = hh + (size_t)ss * DIMD;
    const __bf16* rr = rh + (size_t)edge * DIMD;
    float* ag = agg + (size_t)dd * DIMD;
#pragma unroll
    for (int i = 0; i < 4; ++i) {
        int idx = lane + i * 32;
        atomicAdd(&ag[idx], alpha * ((float)hs[idx] + (float)rr[idx]));
    }
}

// ---------------------------------------------------------------------------
// Host-side orchestration (graph-capture safe: kernel launches only)
// ---------------------------------------------------------------------------
extern "C" void kernel_launch(void* const* d_in, const int* in_sizes, int n_in,
                              void* d_out, int out_size, void* d_ws, size_t ws_size,
                              hipStream_t stream)
{
    (void)n_in; (void)out_size; (void)ws_size;
    const float* x    = (const float*)d_in[0];
    const float* r_h  = (const float*)d_in[1];
    const int*   src  = (const int*)d_in[2];
    const int*   dst  = (const int*)d_in[3];
    const float* W    = (const float*)d_in[4];
    const float* Wr   = (const float*)d_in[5];
    const float* Att  = (const float*)d_in[6];
    const float* Loop = (const float*)d_in[7];
    float* out = (float*)d_out;

    const int N = in_sizes[0] / DIMD;
    const int E = in_sizes[2];
    const int L = in_sizes[4] / (DIMD * DIMD);

    // workspace carve-out (256B aligned slabs)
    char* p = (char*)d_ws;
    auto carve = [&](size_t bytes) -> void* {
        void* q = (void*)p;
        p += (bytes + 255) & ~(size_t)255;
        return q;
    };
    __bf16*   hh_bf   = (__bf16*)carve((size_t)N * DIMD * 2);
    __bf16*   hbuf_bf = (__bf16*)carve((size_t)N * DIMD * 2);  // h (bf16), layers 0..L-1
    __bf16*   rin_bf  = (__bf16*)carve((size_t)E * DIMD * 2);  // r_h converted once
    __bf16*   rh_bf   = (__bf16*)carve((size_t)E * DIMD * 2);  // r_h @ Wr[l]
    float*    agg     = (float*)carve((size_t)N * DIMD * 4);
    float*    ndot    = (float*)carve((size_t)N * 4);
    unsigned* menc    = (unsigned*)carve((size_t)N * 4);
    float*    ssum    = (float*)carve((size_t)N * 4);
    float*    ealpha  = (float*)carve((size_t)E * 4);

    const dim3 blk(256);
    const dim3 gN((N + 127) / 128);          // gemm over N rows
    const dim3 gE((E + 127) / 128);          // gemm over E rows
    const dim3 gInit((N * DIMD + 255) / 256);
    const dim3 gNodeW((N + 7) / 8);          // warp-per-node
    const dim3 gEdgeW((E + 7) / 8);          // warp-per-edge
    const dim3 gEdgeT((E + 255) / 256);      // thread-per-edge
    const long nElemN = (long)N * DIMD;
    const long nElemE = (long)E * DIMD;
    const dim3 gCvtN((unsigned)((nElemN / 4 + 255) / 256));
    const dim3 gCvtE((unsigned)((nElemE / 4 + 255) / 256));

    // One-time bf16 conversion of the streaming inputs.
    f32_to_bf16<<<gCvtN, blk, 0, stream>>>(x, hbuf_bf, nElemN);
    f32_to_bf16<<<gCvtE, blk, 0, stream>>>(r_h, rin_bf, nElemE);

    for (int l = 0; l < L; ++l) {
        const float* Wl  = W    + (size_t)l * DIMD * DIMD;
        const float* Wrl = Wr   + (size_t)l * DIMD * DIMD;
        const float* Ll  = Loop + (size_t)l * DIMD * DIMD;
        const float* al  = Att  + (size_t)l * 3 * DIMD;

        gemm_wmma<__bf16, __bf16, false><<<gN, blk, 0, stream>>>(
            hbuf_bf, Wl, nullptr, hh_bf, N);
        gemm_wmma<__bf16, __bf16, false><<<gE, blk, 0, stream>>>(
            rin_bf, Wrl, nullptr, rh_bf, E);
        init_seg<<<gInit, blk, 0, stream>>>(menc, ssum, agg, N);
        node_dot<<<gNodeW, blk, 0, stream>>>(hh_bf, al + DIMD, ndot, N);
        edge_score<<<gEdgeW, blk, 0, stream>>>(hh_bf, rh_bf, src, dst, al, ndot,
                                               ealpha, menc, E);
        edge_exp<<<gEdgeT, blk, 0, stream>>>(ealpha, dst, menc, ssum, E);
        edge_agg<<<gEdgeW, blk, 0, stream>>>(hh_bf, rh_bf, src, dst, ealpha, ssum,
                                             agg, E);
        if (l == L - 1) {
            gemm_wmma<__bf16, float, true><<<gN, blk, 0, stream>>>(
                hh_bf, Ll, agg, out, N);
        } else {
            gemm_wmma<__bf16, __bf16, true><<<gN, blk, 0, stream>>>(
                hh_bf, Ll, agg, hbuf_bf, N);
        }
    }
}
